// Decoder_SICA_40561671143756
// MI455X (gfx1250) — matrix operand order repaired and checked by reading, not compiled, not run on hardware
//
#include <hip/hip_runtime.h>
#include <hip/hip_bf16.h>

// ---------------------------------------------------------------------------
// gfx1250 async global->LDS staging (ASYNCcnt path), guarded so the file
// still compiles (sync fallback) if the toolchain lacks the builtins.
// ---------------------------------------------------------------------------
#if defined(__gfx1250__) && __has_builtin(__builtin_amdgcn_global_load_async_to_lds_b128) && \
    __has_builtin(__builtin_amdgcn_s_wait_asynccnt)
#define ASYNC_A 1
#else
#define ASYNC_A 0
#endif

// builtin parameter types (from hipcc diagnostic: v4i = vector_size(16) int,
// global src in AS1, LDS dst in AS3)
typedef __attribute__((vector_size(4 * sizeof(int)))) int v4i;
typedef __attribute__((address_space(1))) v4i g_v4i;
typedef __attribute__((address_space(3))) v4i l_v4i;

// ---------------------------------------------------------------------------
// Types for CDNA5 WMMA (wave32): D(f32 16x16) = A(bf16 16x32) x B(bf16 32x16) + C
// ---------------------------------------------------------------------------
typedef __attribute__((ext_vector_type(16))) __bf16 bf16x16;
typedef __attribute__((ext_vector_type(8)))  __bf16 bf16x8;
typedef __attribute__((ext_vector_type(8)))  float  f32x8;

__device__ __forceinline__ bf16x16 cat8(bf16x8 lo, bf16x8 hi) {
  return __builtin_shufflevector(lo, hi, 0,1,2,3,4,5,6,7,8,9,10,11,12,13,14,15);
}

// A-fragment per ISA 7.12.2 (16-bit A 16x32): lane r = lane%16 holds row M=r,
// K elements [kb, kb+8) then [kb+16, kb+24) with kb = (lane/16)*8.
__device__ __forceinline__ bf16x16 load_a_frag(const __bf16* row_base, int kb8) {
  bf16x8 lo = *(const bf16x8*)(row_base + kb8);
  bf16x8 hi = *(const bf16x8*)(row_base + kb8 + 16);
  return cat8(lo, hi);
}

__device__ __forceinline__ f32x8 wmma_bf16(bf16x16 a, bf16x16 b, f32x8 c) {
  return __builtin_amdgcn_wmma_f32_16x16x32_bf16(
      /*neg_a=*/false, a, /*neg_b=*/false, b,
      /*c_mod=*/(short)0, c, /*reuse_a=*/false, /*reuse_b=*/false);
}

// ---------------------------------------------------------------------------
// Kernel: pack OIHW fp32 weights -> [O][K] bf16, K index = (ky*KW+kx)*Cin + ci
// ---------------------------------------------------------------------------
__global__ void pack_weight_kernel(const float* __restrict__ w, __bf16* __restrict__ dst,
                                   int O, int Cin, int KH, int KW, int total) {
  int idx = blockIdx.x * blockDim.x + threadIdx.x;
  if (idx >= total) return;
  int K = KH * KW * Cin;
  int o = idx / K;
  int kidx = idx - o * K;
  int kk = kidx / Cin;
  int ci = kidx - kk * Cin;
  int ky = kk / KW;
  int kx = kk - ky * KW;
  dst[idx] = (__bf16)w[((size_t)(o * Cin + ci) * KH + ky) * KW + kx];
}

// ---------------------------------------------------------------------------
// Kernel: einsum  out[b,n,g,o,h*5+p,w*5+q] = sum_c A[b,(n*16+g)*4+c,h,w]*S[b,g*4+c,o,p,q]
// Output NHWC bf16: x0[b][oh*10+ow][ch], ch = n*48 + g*3 + o (384 channels)
// ---------------------------------------------------------------------------
__global__ void einsum_kernel(const float* __restrict__ A, const float* __restrict__ S,
                              __bf16* __restrict__ x0, int total) {
  int idx = blockIdx.x * blockDim.x + threadIdx.x;
  if (idx >= total) return;
  int ch = idx % 384;
  int pos = (idx / 384) % 100;
  int b = idx / 38400;
  int o = ch % 3;
  int g = (ch / 3) & 15;
  int n = ch / 48;
  int oh = pos / 10, ow = pos - (pos / 10) * 10;
  int h = oh / 5, p = oh - h * 5;
  int w = ow / 5, q = ow - w * 5;
  const float* a = A + (size_t)b * 2048 + ((n * 16 + g) * 4) * 4 + h * 2 + w; // +c*4
  const float* s = S + (((size_t)b * 64 + g * 4) * 3 + o) * 25 + p * 5 + q;   // +c*75
  float acc = 0.f;
#pragma unroll
  for (int c = 0; c < 4; ++c) acc += a[c * 4] * s[c * 75];
  x0[idx] = (__bf16)acc;
}

// ---------------------------------------------------------------------------
// Implicit-GEMM conv + bias + ReLU.
//   act: NHWC bf16 [B*Hin*Win][Cin], wgt: [Cout][K] bf16, K = KH*KW*Cin
//   out: NHWC, bf16 (out_bf) or f32 (out_f32)
// 256 threads = 8 waves. Tile: 128(M) x TN(N), TN = BF*32.
// Wave = 32(M) x BF*16(N) -> 2*BF WMMAs per 32-wide K-step.
// - double-buffered LDS, one barrier per K-step, last step peeled
// - A (weight) tile staged with GLOBAL_LOAD_ASYNC_TO_LDS_B128 (ASYNCcnt)
// - B (im2col) tile: branch-free clamped loads + validity mask (zero padding)
// - tap/channel counters advance with adds only (Cin % 32 == 0), no division
// ---------------------------------------------------------------------------
template <int BF>
__global__ __launch_bounds__(256) void conv_gemm_kernel(
    const __bf16* __restrict__ act, const __bf16* __restrict__ wgt,
    const float* __restrict__ bias, __bf16* __restrict__ out_bf,
    float* __restrict__ out_f32,
    int Cin, int Cout, int Hin, int Win, int Hout, int Wout,
    int stride, int pad, int KH, int KW, int K, int Nn) {
  constexpr int TN  = BF * 32;      // block N tile
  constexpr int HPT = TN / 8;       // B halves per thread (16 or 8)
  constexpr int TPC = 32 / HPT;     // threads per B column (2 or 4)
  constexpr int NU4 = HPT / 8;      // uint4 loads per thread for B (2 or 1)

  __shared__ alignas(64) __bf16 As[2][128 * 32]; // [m][k]
  __shared__ alignas(64) __bf16 Bs[2][TN * 32];  // [n][k]

  const int tid  = threadIdx.x;
  const int lane = tid & 31;
  const int wid  = tid >> 5;
  const int wm   = wid & 3;   // 0..3 -> M sub-tile of 32
  const int wn   = wid >> 2;  // 0..1 -> N sub-tile of BF*16
  const int m_tile = blockIdx.y * 128;
  const int n_tile = blockIdx.x * TN;

  // ---- A loader: row (tid&127), K halves [aseg*16, aseg*16+16) ----
  const int ma   = tid & 127;
  const int aseg = tid >> 7;
  const __bf16* aA = wgt + (size_t)(m_tile + ma) * K + aseg * 16; // +32/blk
  const int a_lds = ma * 32 + aseg * 16;

  // ---- B loader: column tid/TPC, K halves [kseg, kseg+HPT) ----
  const int nb_local = tid / TPC;
  const int kseg     = (tid % TPC) * HPT;
  const int ng = n_tile + nb_local;        // always < Nn (tiles divide exactly)
  const int HW = Hout * Wout;
  int b_img = ng / HW;
  int rem = ng - b_img * HW;
  int oh = rem / Wout;
  int ow = rem - oh * Wout;
  const int hbase = oh * stride - pad;
  const int wbase = ow * stride - pad;
  const __bf16* act_img = act + (size_t)b_img * Hin * Win * Cin;
  const int b_lds = nb_local * 32 + kseg;

  // ---- compute-side fragment offsets (halves, buffer-relative) ----
  const int lr   = lane & 15;
  const int kb8  = (lane >> 4) * 8;
  const int kb16 = (lane >> 4) * 16;
  const int a_off0 = (wm * 32 + lr) * 32;
  const int a_off1 = a_off0 + 16 * 32;
  int b_off[BF];
#pragma unroll
  for (int j = 0; j < BF; ++j) b_off[j] = (wn * BF * 16 + j * 16 + lr) * 32 + kb8;

  f32x8 acc[2][BF] = {};

  // ---- prefetch tap/channel counters (uniform; adds only) ----
  int pc0 = 0, pky = 0, pkx = 0;
  uint4 av0, av1, bv[NU4];

  // A tile: global -> LDS (async on gfx1250, sync fallback otherwise)
  auto a_fetch = [&](int buf) {
#if ASYNC_A
    // inst offset applies to BOTH addresses, so one base pair covers the two
    // 16B chunks of this thread's 32B slice.
    __builtin_amdgcn_global_load_async_to_lds_b128(
        (g_v4i*)aA, (l_v4i*)&As[buf][a_lds], 0, 0);
    __builtin_amdgcn_global_load_async_to_lds_b128(
        (g_v4i*)aA, (l_v4i*)&As[buf][a_lds], 16, 0);
#else
    const uint4* g = (const uint4*)aA;
    av0 = g[0];
    av1 = g[1];
#endif
    aA += 32;
  };
  auto a_commit = [&](int buf) {
#if !ASYNC_A
    uint4* al = (uint4*)&As[buf][a_lds];
    al[0] = av0;
    al[1] = av1;
#endif
    (void)buf;
  };

  // B tile: branch-free. Clamp tap coords in range, always load, then AND
  // with 0/~0 validity mask (implicit-GEMM zero padding without exec dance).
  auto b_fetch = [&]() {
    const int hin = hbase + pky;
    const int win = wbase + pkx;
    const bool v = ((unsigned)hin < (unsigned)Hin) &&
                   ((unsigned)win < (unsigned)Win);
    const int hc = min(max(hin, 0), Hin - 1);
    const int wc = min(max(win, 0), Win - 1);
    const __bf16* src = act_img + ((size_t)hc * Win + wc) * Cin + pc0 + kseg;
    const unsigned msk = v ? 0xFFFFFFFFu : 0u;
#pragma unroll
    for (int u = 0; u < NU4; ++u) {
      uint4 t = ((const uint4*)src)[u];
      t.x &= msk; t.y &= msk; t.z &= msk; t.w &= msk;
      bv[u] = t;
    }
    pc0 += 32;                          // advance tap/channel counters
    if (pc0 == Cin) {
      pc0 = 0;
      if (++pkx == KW) { pkx = 0; ++pky; }
    }
  };
  auto b_commit = [&](int buf) {
    uint4* blp = (uint4*)&Bs[buf][b_lds];
#pragma unroll
    for (int u = 0; u < NU4; ++u) blp[u] = bv[u];
  };

  auto wait_stage = [&]() {
#if ASYNC_A
    __builtin_amdgcn_s_wait_asynccnt(0);  // own async LDS writes landed
#endif
    __syncthreads();                      // ...and visible to all waves
  };

  auto compute = [&](int cur) {
    const __bf16* Ab = As[cur];
    const __bf16* Bb = Bs[cur];
    bf16x16 a0 = load_a_frag(Ab + a_off0, kb8);
    bf16x16 a1 = load_a_frag(Ab + a_off1, kb8);
    bf16x16 bf[BF];
#pragma unroll
    for (int j = 0; j < BF; ++j) bf[j] = load_a_frag(Bb + b_off[j], 0);
#pragma unroll
    for (int j = 0; j < BF; ++j) acc[0][j] = wmma_bf16(a0, bf[j], acc[0][j]);
#pragma unroll
    for (int j = 0; j < BF; ++j) acc[1][j] = wmma_bf16(a1, bf[j], acc[1][j]);
  };

  const int kblocks = K >> 5;

  // prologue: stage K-block 0 into buffer 0
  a_fetch(0);
  b_fetch();
  a_commit(0);
  b_commit(0);

  // steady state: branch-free (last K-step peeled)
  for (int kb = 0; kb < kblocks - 1; ++kb) {
    wait_stage();                        // buf (kb&1) ready for all waves
    const int cur = kb & 1;
    const int nxt = cur ^ 1;
    a_fetch(nxt);                        // async/global loads overlap WMMAs
    b_fetch();
    compute(cur);
    a_commit(nxt);
    b_commit(nxt);
  }
  wait_stage();
  compute((kblocks - 1) & 1);

  // ---- epilogue: bias + ReLU, NHWC store (8 channels contiguous per lane) ----
#pragma unroll
  for (int ii = 0; ii < 2; ++ii) {
#pragma unroll
    for (int jj = 0; jj < BF; ++jj) {
      const int ncol = n_tile + wn * BF * 16 + jj * 16 + lr;
      const int mrow = m_tile + wm * 32 + ii * 16 + (lane >> 4) * 8;
      if (ncol < Nn) {
        f32x8 r = acc[ii][jj];
        if (out_f32 != nullptr) {
          float* dst = out_f32 + (size_t)ncol * Cout + mrow;
#pragma unroll
          for (int g = 0; g < 8; ++g) dst[g] = fmaxf(r[g] + bias[mrow + g], 0.0f);
        } else {
          union { __bf16 h[8]; uint4 q; } pk;
#pragma unroll
          for (int g = 0; g < 8; ++g)
            pk.h[g] = (__bf16)fmaxf(r[g] + bias[mrow + g], 0.0f);
          *(uint4*)(out_bf + (size_t)ncol * Cout + mrow) = pk.q;
        }
      }
    }
  }
}

// ---------------------------------------------------------------------------
// Kernel: linear 512 -> 10, one wave32 per batch row, shuffle reduction.
// ---------------------------------------------------------------------------
__global__ __launch_bounds__(32) void linear_kernel(const float* __restrict__ x,
                                                    const float* __restrict__ Wl,
                                                    const float* __restrict__ bl,
                                                    float* __restrict__ out) {
  const int b = blockIdx.x;
  const int lane = threadIdx.x;
  const float* xb = x + (size_t)b * 512;
  float acc[10];
#pragma unroll
  for (int j = 0; j < 10; ++j) acc[j] = 0.f;
  for (int k = lane; k < 512; k += 32) {
    float xv = xb[k];
#pragma unroll
    for (int j = 0; j < 10; ++j) acc[j] += xv * Wl[j * 512 + k];
  }
#pragma unroll
  for (int j = 0; j < 10; ++j) {
#pragma unroll
    for (int off = 16; off > 0; off >>= 1) acc[j] += __shfl_down(acc[j], off, 32);
  }
  if (lane == 0) {
#pragma unroll
    for (int j = 0; j < 10; ++j) out[b * 10 + j] = acc[j] + bl[j];
  }
}

// ---------------------------------------------------------------------------
// Host launcher
// ---------------------------------------------------------------------------
extern "C" void kernel_launch(void* const* d_in, const int* in_sizes, int n_in,
                              void* d_out, int out_size, void* d_ws, size_t ws_size,
                              hipStream_t stream) {
  (void)in_sizes; (void)n_in; (void)out_size; (void)ws_size;
  const float* A  = (const float*)d_in[0];
  const float* S  = (const float*)d_in[1];
  const float* W1 = (const float*)d_in[2];  const float* b1 = (const float*)d_in[3];
  const float* W2 = (const float*)d_in[4];  const float* b2 = (const float*)d_in[5];
  const float* W3 = (const float*)d_in[6];  const float* b3 = (const float*)d_in[7];
  const float* W4 = (const float*)d_in[8];  const float* b4 = (const float*)d_in[9];
  const float* Wl = (const float*)d_in[10]; const float* bl = (const float*)d_in[11];
  float* out = (float*)d_out;

  const int B = 512;

  // ---- carve workspace ----
  char* ws = (char*)d_ws;
  size_t off = 0;
  auto carve = [&](size_t bytes) {
    char* p = ws + off;
    off += (bytes + 255) & ~(size_t)255;
    return p;
  };
  __bf16* x0  = (__bf16*)carve((size_t)B * 100 * 384 * 2); // (B,10,10,384) NHWC
  __bf16* x1  = (__bf16*)carve((size_t)B * 100 * 384 * 2); // (B,10,10,384)
  __bf16* x2  = (__bf16*)carve((size_t)B * 25 * 384 * 2);  // (B,5,5,384)
  __bf16* x3  = (__bf16*)carve((size_t)B * 25 * 512 * 2);  // (B,5,5,512)
  float*  x4  = (float*)carve((size_t)B * 512 * 4);        // (B,512)
  __bf16* w1b = (__bf16*)carve((size_t)384 * 3456 * 2);
  __bf16* w2b = (__bf16*)carve((size_t)384 * 3456 * 2);
  __bf16* w3b = (__bf16*)carve((size_t)512 * 3456 * 2);
  __bf16* w4b = (__bf16*)carve((size_t)512 * 12800 * 2);

  // ---- pack weights fp32 OIHW -> bf16 [O][(ky*KW+kx)*Cin+ci] ----
  {
    int t1 = 384 * 3456;
    pack_weight_kernel<<<(t1 + 255) / 256, 256, 0, stream>>>(W1, w1b, 384, 384, 3, 3, t1);
    pack_weight_kernel<<<(t1 + 255) / 256, 256, 0, stream>>>(W2, w2b, 384, 384, 3, 3, t1);
    int t3 = 512 * 3456;
    pack_weight_kernel<<<(t3 + 255) / 256, 256, 0, stream>>>(W3, w3b, 512, 384, 3, 3, t3);
    int t4 = 512 * 12800;
    pack_weight_kernel<<<(t4 + 255) / 256, 256, 0, stream>>>(W4, w4b, 512, 512, 5, 5, t4);
  }

  // ---- einsum -> x0 (NHWC bf16) ----
  {
    int total = B * 100 * 384;
    einsum_kernel<<<(total + 255) / 256, 256, 0, stream>>>(A, S, x0, total);
  }

  // ---- conv1: 384->384, 10x10, 3x3 s1 p1 ----
  {
    int N = B * 100;
    dim3 grid(N / 128, 384 / 128);
    conv_gemm_kernel<4><<<grid, 256, 0, stream>>>(x0, w1b, b1, x1, nullptr,
        384, 384, 10, 10, 10, 10, 1, 1, 3, 3, 3456, N);
  }
  // ---- conv2: 384->384, 10x10 -> 5x5, 3x3 s2 p1 ----
  {
    int N = B * 25;
    dim3 grid(N / 128, 384 / 128);
    conv_gemm_kernel<4><<<grid, 256, 0, stream>>>(x1, w2b, b2, x2, nullptr,
        384, 384, 10, 10, 5, 5, 2, 1, 3, 3, 3456, N);
  }
  // ---- conv3: 384->512, 5x5, 3x3 s1 p1 ----
  {
    int N = B * 25;
    dim3 grid(N / 128, 512 / 128);
    conv_gemm_kernel<4><<<grid, 256, 0, stream>>>(x2, w3b, b3, x3, nullptr,
        384, 512, 5, 5, 5, 5, 1, 1, 3, 3, 3456, N);
  }
  // ---- conv4: 512->512, 5x5 -> 1x1, 5x5 s1 p0 (f32 out, narrow N tile) ----
  {
    int N = B;
    dim3 grid(N / 64, 512 / 128);
    conv_gemm_kernel<2><<<grid, 256, 0, stream>>>(x3, w4b, b4, nullptr, x4,
        512, 512, 5, 5, 1, 1, 1, 0, 5, 5, 12800, N);
  }
  // ---- linear 512 -> 10 ----
  linear_kernel<<<B, 32, 0, stream>>>(x4, Wl, bl, out);
}
